// ModulatedConv2d_85203561218719
// MI455X (gfx1250) — matrix operand order repaired and checked by reading, not compiled, not run on hardware
//
#include <hip/hip_runtime.h>
#include <hip/hip_bf16.h>

typedef __attribute__((ext_vector_type(16))) __bf16 v16bf;
typedef __attribute__((ext_vector_type(8)))  __bf16 v8bf;
typedef __attribute__((ext_vector_type(8)))  float  v8f;
typedef __attribute__((ext_vector_type(4)))  float  v4f;

#define B_   8
#define CI_  128
#define CO_  128
#define SD_  512
#define H_   256
#define W_   256

// Weights pre-packed in exact WMMA A-fragment order:
// [b][co_strip(8)][ci_chunk(4)][tap(9)][lane(32)][elem(16)]  (bf16)
__device__ __align__(32) __bf16 g_wpack[(size_t)B_ * 8 * 4 * 9 * 32 * 16];
__device__ float g_s[B_ * CI_];

// ---------------- Kernel 1: affine style s = style @ (aff_w*gain)^T + aff_b --
__global__ void style_affine_kernel(const float* __restrict__ style,
                                    const float* __restrict__ aff_w,
                                    const float* __restrict__ aff_b) {
  int b  = blockIdx.x >> 7;   // 1024 blocks = b*128 + ci
  int ci = blockIdx.x & 127;
  int t  = threadIdx.x;       // 128 threads
  const float gain = 0.04419417382415922f;  // 1/sqrt(512)
  float p = 0.f;
#pragma unroll
  for (int j = 0; j < 4; ++j) {
    int k = t + j * 128;
    p += style[b * SD_ + k] * aff_w[ci * SD_ + k];
  }
  __shared__ float red[128];
  red[t] = p;
  __syncthreads();
  for (int off = 64; off > 0; off >>= 1) {
    if (t < off) red[t] += red[t + off];
    __syncthreads();
  }
  if (t == 0) g_s[b * CI_ + ci] = red[0] * gain + aff_b[ci];
}

// ---------------- Kernel 2: modulate + demodulate + pack into WMMA A layout --
__global__ void modpack_kernel(const float* __restrict__ w_base) {
  int b  = blockIdx.x >> 7;   // 1024 blocks = b*128 + co
  int co = blockIdx.x & 127;
  int ci = threadIdx.x;       // 128 threads, one per ci
  const float wgain = 0.029462782549439483f;  // 1/sqrt(128*9)
  float s = g_s[b * CI_ + ci] * wgain;
  const float* wb = w_base + ((size_t)co * CI_ + ci) * 9;
  float v[9];
  float sq = 0.f;
#pragma unroll
  for (int tap = 0; tap < 9; ++tap) { v[tap] = wb[tap] * s; sq += v[tap] * v[tap]; }
  __shared__ float red[128];
  __shared__ float dsh;
  red[ci] = sq;
  __syncthreads();
  for (int off = 64; off > 0; off >>= 1) {
    if (ci < off) red[ci] += red[ci + off];
    __syncthreads();
  }
  if (ci == 0) dsh = rsqrtf(red[0] + 1e-8f);
  __syncthreads();
  float d = dsh;

  // Invert the 16-bit A-matrix (16x32) VGPR layout: lane = half*16 + M,
  // K = (e/8)*16 + half*8 + (e%8).
  int co_strip = co >> 4, m = co & 15;
  int cc = ci >> 5, cl = ci & 31;
  int hi = cl >> 3;
  int half = hi & 1;
  int e = ((hi >> 1) << 3) | (cl & 7);
  int lane = half * 16 + m;
#pragma unroll
  for (int tap = 0; tap < 9; ++tap) {
    size_t off = (((((size_t)b * 8 + co_strip) * 4 + cc) * 9 + tap) * 32 + lane) * 16 + e;
    g_wpack[off] = (__bf16)(v[tap] * d);
  }
}

// Stage one ci-chunk (32 ci x 3 rows x [1 + 128 + 3] cols) into an LDS buffer.
// Interior 128 cols are always in-range in w -> vectorized b128 path;
// halo cols (c = 0, 129..131) take the scalar bounds-checked path.
__device__ __forceinline__ void stage_chunk(const float* __restrict__ x,
                                            int b, int cc, int h, int w0, int tid,
                                            __bf16 (*dst)[132][40]) {
  // interior: 96 rows (32 ci x 3 r) x 32 segments of 4 floats = 3072 v4f
#pragma unroll
  for (int k = 0; k < 12; ++k) {
    int idx = tid + (k << 8);      // 0..3071; one row per wave per k
    int row = idx >> 5;            // 0..95
    int seg = idx & 31;            // 0..31
    int cl  = row / 3;
    int r   = row - cl * 3;
    int hg  = h + r - 1;
    v4f val = {0.f, 0.f, 0.f, 0.f};
    if ((unsigned)hg < H_)
      val = *reinterpret_cast<const v4f*>(
          &x[(((size_t)b * CI_ + cc * 32 + cl) * H_ + hg) * W_ + w0 + seg * 4]);
    int c0 = 1 + seg * 4;
#pragma unroll
    for (int i = 0; i < 4; ++i) dst[r][c0 + i][cl] = (__bf16)val[i];
  }
  // halo: 4 cols x 96 rows = 384 scalar elements
  for (int idx = tid; idx < 384; idx += 256) {
    int colid = idx & 3;
    int row   = idx >> 2;          // 0..95
    int cl = row / 3;
    int r  = row - cl * 3;
    int c  = (colid == 0) ? 0 : (128 + colid);  // 0, 129, 130, 131
    int hg = h + r - 1;
    int wg = w0 + c - 1;
    float val = 0.f;
    if ((unsigned)hg < H_ && (unsigned)wg < W_)
      val = x[(((size_t)b * CI_ + cc * 32 + cl) * H_ + hg) * W_ + wg];
    dst[r][c][cl] = (__bf16)val;
  }
}

// ---------------- Kernel 3: implicit-GEMM conv via bf16 WMMA -----------------
// Block: one (b, h, half-row of 128 px) x all 128 CO. 256 thr = 8 waves.
// Wave wv owns CO rows [16*wv, 16*wv+16); 8 accumulator tiles of 16x16 cover
// the 128 pixels. K loop = 4 ci-chunks of 32 x 9 taps, double-buffered LDS;
// staging of chunk cc+1 runs after compute of cc (cross-wave overlap, single
// barrier per iteration).
__global__ __launch_bounds__(256) void modconv_kernel(const float* __restrict__ x,
                                                      float* __restrict__ y) {
  int bid = blockIdx.x;          // 4096 = B * 256 rows * 2 half-rows
  int b   = bid >> 9;
  int t   = bid & 511;
  int h   = t >> 1;
  int w0  = (t & 1) << 7;
  int tid  = threadIdx.x;
  int wv   = tid >> 5;
  int lane = tid & 31;
  int lm   = lane & 15;
  int half = lane >> 4;

  // double-buffered x tile, bf16: [buf 2][row 3][col 132][ci 32 pad->40]
  __shared__ __bf16 xs[2][3][132][40];   // 63.4 KB

  stage_chunk(x, b, 0, h, w0, tid, xs[0]);
  __syncthreads();

  v8f acc[8] = {};

  for (int cc = 0; cc < 4; ++cc) {
    int cur = cc & 1;

    // ---- compute: 9 taps x 8 N-tiles of WMMA on buffer `cur`
    const v16bf* wbase = reinterpret_cast<const v16bf*>(g_wpack) +
                         ((((size_t)b * 8 + wv) * 4 + cc) * 9) * 32 + lane;
#pragma unroll
    for (int tap = 0; tap < 9; ++tap) {
      int kh = tap / 3, kw = tap % 3;
      v16bf af = wbase[tap * 32];
      v16bf bfr[8];
#pragma unroll
      for (int nt = 0; nt < 8; ++nt) {
        // B fragment (32x16 bf16): N = lane%16, K = (e/8)*16 + half*8 + (e%8)
        // => two runs of 8 ci values at fixed (row, col) in LDS.
        int c = nt * 16 + lm + kw;
        const __bf16* bp = &xs[cur][kh][c][half * 8];
        v8bf lo  = *reinterpret_cast<const v8bf*>(bp);
        v8bf hiv = *reinterpret_cast<const v8bf*>(bp + 16);
        bfr[nt] = __builtin_shufflevector(lo, hiv,
            0, 1, 2, 3, 4, 5, 6, 7, 8, 9, 10, 11, 12, 13, 14, 15);
      }
#pragma unroll
      for (int nt = 0; nt < 8; ++nt)
        acc[nt] = __builtin_amdgcn_wmma_f32_16x16x32_bf16(
            false, af, false, bfr[nt], (short)0, acc[nt], false, false);
    }

    // ---- stage chunk cc+1 into the other buffer (overlaps across waves)
    if (cc < 3) stage_chunk(x, b, cc + 1, h, w0, tid, xs[cur ^ 1]);
    __syncthreads();
  }

  // C/D layout: M = vgpr + 8*half, N = lane%16. Non-temporal stores: y is
  // write-once; keep L2 for x halo reuse.
  size_t ybase = ((size_t)b * CO_ + wv * 16) * (H_ * W_) + (size_t)h * W_ + w0;
#pragma unroll
  for (int nt = 0; nt < 8; ++nt) {
#pragma unroll
    for (int v = 0; v < 8; ++v) {
      int m = v + 8 * half;
      int n = nt * 16 + lm;
      __builtin_nontemporal_store(acc[nt][v], &y[ybase + (size_t)m * (H_ * W_) + n]);
    }
  }
}

extern "C" void kernel_launch(void* const* d_in, const int* in_sizes, int n_in,
                              void* d_out, int out_size, void* d_ws, size_t ws_size,
                              hipStream_t stream) {
  (void)in_sizes; (void)n_in; (void)d_ws; (void)ws_size; (void)out_size;
  const float* x      = (const float*)d_in[0];
  const float* style  = (const float*)d_in[1];
  const float* w_base = (const float*)d_in[2];
  const float* aff_w  = (const float*)d_in[3];
  const float* aff_b  = (const float*)d_in[4];
  float* y = (float*)d_out;

  style_affine_kernel<<<dim3(B_ * CI_), dim3(128), 0, stream>>>(style, aff_w, aff_b);
  modpack_kernel<<<dim3(B_ * CO_), dim3(128), 0, stream>>>(w_base);
  modconv_kernel<<<dim3(B_ * (H_ * W_ / 128)), dim3(256), 0, stream>>>(x, y);
}
